// TopDownCore11_24232205484550
// MI455X (gfx1250) — compile-verified, matrix operand order checked by reading, not run on hardware
//
#include <hip/hip_runtime.h>
#include <math.h>

// ---------------- types ----------------
typedef __attribute__((ext_vector_type(16))) __bf16 v16bf;
typedef __attribute__((ext_vector_type(8)))  __bf16 v8bf_t;
typedef __attribute__((ext_vector_type(8)))  float  v8f;

#define Bn 256
#define Ln 196
#define Rn 1024
#define En 1024
#define Hn 512

// ---------------- f32 -> bf16 strided slice convert ----------------
__global__ void cvt_slice_k(__bf16* __restrict__ dst, int dld,
                            const float* __restrict__ src, int sld,
                            int rows, int cols) {
    int idx = blockIdx.x * blockDim.x + threadIdx.x;
    if (idx >= rows * cols) return;
    int r = idx / cols;
    int c = idx - r * cols;
    dst[(size_t)r * dld + c] = (__bf16)src[(size_t)r * sld + c];
}

// ---------------- bf16 fragment loader (16-bit A/B WMMA layout) ----------------
// Lane m (0-15): elems 0..7 = K[k0..k0+7],   elems 8..15 = K[k0+16..k0+23]
// Lane m+16   : elems 0..7 = K[k0+8..k0+15], elems 8..15 = K[k0+24..k0+31]
__device__ __forceinline__ v16bf load_frag(const __bf16* __restrict__ p,
                                           int ld, int row, int k0, int lane) {
    const __bf16* q = p + (size_t)row * ld + k0 + ((lane & 16) ? 8 : 0);
    v8bf_t lo = *(const v8bf_t*)q;        // 16B global_load_b128
    v8bf_t hi = *(const v8bf_t*)(q + 16); // 16B global_load_b128
    v16bf r;
#pragma unroll
    for (int i = 0; i < 8; ++i) { r[i] = lo[i]; r[i + 8] = hi[i]; }
    return r;
}

#define WMMA_BF16(a, b, c) \
    __builtin_amdgcn_wmma_f32_16x16x32_bf16(false, (a), false, (b), (short)0, (c), false, false)

// K-loop, unrolled x2 with two named fragment sets that ping-pong by register
// ASSIGNMENT (no cur=next copies -> no v_dual_mov / v_nop hazard filler).
// Per 64-K iteration: load set1(k+32), 8 WMMA on set0, load set0(k+64), 8 WMMA
// on set1. Loads stay in flight behind the matrix math. Requires K % 64 == 0.
__device__ __forceinline__ void mma_loop(const __bf16* __restrict__ A,
                                         const __bf16* __restrict__ W,
                                         int K, int ar0, int ar1,
                                         int nb0, int lane, v8f acc[2][4]) {
    v16bf a0 = load_frag(A, K, ar0, 0, lane);
    v16bf a1 = load_frag(A, K, ar1, 0, lane);
    v16bf b0 = load_frag(W, K, nb0 +  0, 0, lane);
    v16bf b1 = load_frag(W, K, nb0 + 16, 0, lane);
    v16bf b2 = load_frag(W, K, nb0 + 32, 0, lane);
    v16bf b3 = load_frag(W, K, nb0 + 48, 0, lane);
    for (int k0 = 0; k0 < K; k0 += 64) {
        int k1 = k0 + 32;
        v16bf c0 = load_frag(A, K, ar0, k1, lane);
        v16bf c1 = load_frag(A, K, ar1, k1, lane);
        v16bf d0 = load_frag(W, K, nb0 +  0, k1, lane);
        v16bf d1 = load_frag(W, K, nb0 + 16, k1, lane);
        v16bf d2 = load_frag(W, K, nb0 + 32, k1, lane);
        v16bf d3 = load_frag(W, K, nb0 + 48, k1, lane);

        acc[0][0] = WMMA_BF16(a0, b0, acc[0][0]);
        acc[1][0] = WMMA_BF16(a1, b0, acc[1][0]);
        acc[0][1] = WMMA_BF16(a0, b1, acc[0][1]);
        acc[1][1] = WMMA_BF16(a1, b1, acc[1][1]);
        acc[0][2] = WMMA_BF16(a0, b2, acc[0][2]);
        acc[1][2] = WMMA_BF16(a1, b2, acc[1][2]);
        acc[0][3] = WMMA_BF16(a0, b3, acc[0][3]);
        acc[1][3] = WMMA_BF16(a1, b3, acc[1][3]);

        int k2 = k0 + 64;
        if (k2 < K) {
            a0 = load_frag(A, K, ar0, k2, lane);
            a1 = load_frag(A, K, ar1, k2, lane);
            b0 = load_frag(W, K, nb0 +  0, k2, lane);
            b1 = load_frag(W, K, nb0 + 16, k2, lane);
            b2 = load_frag(W, K, nb0 + 32, k2, lane);
            b3 = load_frag(W, K, nb0 + 48, k2, lane);
        }

        acc[0][0] = WMMA_BF16(c0, d0, acc[0][0]);
        acc[1][0] = WMMA_BF16(c1, d0, acc[1][0]);
        acc[0][1] = WMMA_BF16(c0, d1, acc[0][1]);
        acc[1][1] = WMMA_BF16(c1, d1, acc[1][1]);
        acc[0][2] = WMMA_BF16(c0, d2, acc[0][2]);
        acc[1][2] = WMMA_BF16(c1, d2, acc[1][2]);
        acc[0][3] = WMMA_BF16(c0, d3, acc[0][3]);
        acc[1][3] = WMMA_BF16(c1, d3, acc[1][3]);
    }
}

// ---------------- bf16 WMMA GEMM:  C[M,N] = A1@W1^T + A2@W2^T + b1 + b2
// M fixed = 256. Block = 8 waves; wave tile 32x64; block tile 64x256.
__global__ void __launch_bounds__(256) gemm_wmma_k(
    const __bf16* __restrict__ A1, const __bf16* __restrict__ W1, int K1,
    const __bf16* __restrict__ A2, const __bf16* __restrict__ W2, int K2,
    const float* __restrict__ bias1, const float* __restrict__ bias2,
    float* __restrict__ C, int N) {
    int lane = threadIdx.x & 31;
    int wave = threadIdx.x >> 5;
    int wn = wave & 3, wm = wave >> 2;          // wm in 0..1
    int m0 = blockIdx.y * 64 + wm * 32;         // two 16-row fragments
    int n0 = blockIdx.x * 256 + wn * 64;
    int ar0 = m0 + (lane & 15);
    int ar1 = m0 + 16 + (lane & 15);
    int nb0 = n0 + (lane & 15);

    v8f acc[2][4];
#pragma unroll
    for (int m = 0; m < 2; ++m)
#pragma unroll
        for (int t = 0; t < 4; ++t)
#pragma unroll
            for (int i = 0; i < 8; ++i) acc[m][t][i] = 0.0f;

    mma_loop(A1, W1, K1, ar0, ar1, nb0, lane, acc);
    if (K2 > 0) mma_loop(A2, W2, K2, ar0, ar1, nb0, lane, acc);

#pragma unroll
    for (int m = 0; m < 2; ++m) {
        int rbase = m0 + m * 16 + ((lane & 16) ? 8 : 0);
#pragma unroll
        for (int t = 0; t < 4; ++t) {
            int col = n0 + t * 16 + (lane & 15);
            float bs = (bias1 ? bias1[col] : 0.0f) + (bias2 ? bias2[col] : 0.0f);
#pragma unroll
            for (int r = 0; r < 8; ++r)
                C[(size_t)(rbase + r) * N + col] = acc[m][t][r] + bs;
        }
    }
}

// ---------------- LSTM pointwise: gates(B,4R) -> h,c ----------------
__global__ void lstm_pointwise_k(const float* __restrict__ gates,
                                 const float* __restrict__ c_prev,
                                 float* __restrict__ h_out,
                                 float* __restrict__ c_out,
                                 float* __restrict__ h_out2) {
    int idx = blockIdx.x * 256 + threadIdx.x;   // [0, B*R)
    int b = idx >> 10;                          // R = 1024
    int j = idx & 1023;
    const float* g = gates + (size_t)b * 4096;
    float gi = g[j], gf = g[1024 + j], gg = g[2048 + j], go = g[3072 + j];
    float si = 1.0f / (1.0f + expf(-gi));
    float sf = 1.0f / (1.0f + expf(-gf));
    float so = 1.0f / (1.0f + expf(-go));
    float c  = sf * c_prev[idx] + si * tanhf(gg);
    float h  = so * tanhf(c);
    h_out[idx] = h;
    c_out[idx] = c;
    if (h_out2) h_out2[idx] = h;
}

// ---------------- attention scores: one wave per (b,l) ----------------
__global__ void att_scores_k(const float* __restrict__ p_att,
                             const float* __restrict__ att_h,
                             const float* __restrict__ Wa,
                             const float* __restrict__ ba,
                             float* __restrict__ scores) {
    int lane = threadIdx.x & 31;
    int wv = blockIdx.x * 8 + (threadIdx.x >> 5);   // (b*L + l)
    int b = wv / Ln;
    const float* p  = p_att + (size_t)wv * Hn;
    const float* ah = att_h + (size_t)b * Hn;
    float s = 0.0f;
    for (int h = lane; h < Hn; h += 32)
        s += tanhf(p[h] + ah[h]) * Wa[h];
#pragma unroll
    for (int off = 16; off; off >>= 1) s += __shfl_xor(s, off, 32);
    if (lane == 0) scores[wv] = s + ba[0];
}

// ---------------- masked softmax over L per row b ----------------
__global__ void softmax_mask_k(const float* __restrict__ scores,
                               const float* __restrict__ mask,
                               float* __restrict__ w) {
    __shared__ float red[256];
    int b = blockIdx.x, tid = threadIdx.x;
    float v = (tid < Ln) ? scores[b * Ln + tid] : -3.4e38f;
    red[tid] = v; __syncthreads();
    for (int s = 128; s; s >>= 1) { if (tid < s) red[tid] = fmaxf(red[tid], red[tid + s]); __syncthreads(); }
    float vmax = red[0]; __syncthreads();
    float e = (tid < Ln) ? expf(v - vmax) : 0.0f;
    red[tid] = e; __syncthreads();
    for (int s = 128; s; s >>= 1) { if (tid < s) red[tid] += red[tid + s]; __syncthreads(); }
    float denom = red[0]; __syncthreads();
    float wm = (tid < Ln) ? (e / denom) * mask[b * Ln + tid] : 0.0f;
    red[tid] = wm; __syncthreads();
    for (int s = 128; s; s >>= 1) { if (tid < s) red[tid] += red[tid + s]; __syncthreads(); }
    float s2 = red[0];
    if (tid < Ln) w[b * Ln + tid] = wm / (s2 + 1e-10f);
}

// ---------------- weighted sum over L: out[b,d] = sum_l w[b,l]*feats[b,l,d] ----------------
__global__ void wsum_k(const float* __restrict__ w,
                       const float* __restrict__ feats,
                       float* __restrict__ out) {
    __shared__ float wsh[Ln];
    int b = blockIdx.y;
    int d = blockIdx.x * 256 + threadIdx.x;
    if (threadIdx.x < Ln) wsh[threadIdx.x] = w[b * Ln + threadIdx.x];
    __syncthreads();
    const float* base = feats + ((size_t)b * Ln) * Rn + d;
    float acc = 0.0f;
#pragma unroll 4
    for (int l = 0; l < Ln; ++l) acc += wsh[l] * base[(size_t)l * Rn];
    out[(size_t)b * Rn + d] = acc;
}

// ---------------- host launcher ----------------
extern "C" void kernel_launch(void* const* d_in, const int* in_sizes, int n_in,
                              void* d_out, int out_size, void* d_ws, size_t ws_size,
                              hipStream_t stream) {
    (void)in_sizes; (void)n_in; (void)out_size; (void)ws_size;
    const size_t BR = (size_t)Bn * Rn;

    const float* xt        = (const float*)d_in[0];
    const float* fc        = (const float*)d_in[1];
    const float* att_feats = (const float*)d_in[2];
    const float* p_att     = (const float*)d_in[3];
    const float* mask      = (const float*)d_in[4];
    const float* att_feats1= (const float*)d_in[5];
    const float* p_att1    = (const float*)d_in[6];
    const float* mask1     = (const float*)d_in[7];
    const float* state_h   = (const float*)d_in[8];
    const float* state_c   = (const float*)d_in[9];
    const float* Wih0 = (const float*)d_in[10]; const float* Whh0 = (const float*)d_in[11];
    const float* bih0 = (const float*)d_in[12]; const float* bhh0 = (const float*)d_in[13];
    const float* Wih1 = (const float*)d_in[14]; const float* Whh1 = (const float*)d_in[15];
    const float* bih1 = (const float*)d_in[16]; const float* bhh1 = (const float*)d_in[17];
    const float* WihL = (const float*)d_in[18]; const float* WhhL = (const float*)d_in[19];
    const float* bihL = (const float*)d_in[20]; const float* bhhL = (const float*)d_in[21];
    const float* a0_Wh = (const float*)d_in[22]; const float* a0_bh = (const float*)d_in[23];
    const float* a0_Wa = (const float*)d_in[24]; const float* a0_ba = (const float*)d_in[25];
    const float* a_Wh  = (const float*)d_in[26]; const float* a_bh  = (const float*)d_in[27];
    const float* a_Wa  = (const float*)d_in[28]; const float* a_ba  = (const float*)d_in[29];
    const float* a1_Wh = (const float*)d_in[30]; const float* a1_bh = (const float*)d_in[31];
    const float* a1_Wa = (const float*)d_in[32]; const float* a1_ba = (const float*)d_in[33];

    float* out   = (float*)d_out;      // output (B,R)
    float* out_h = out + BR;           // new_h (3,B,R)
    float* out_c = out + 4 * BR;       // new_c (3,B,R)

    // workspace layout (all offsets 256B-aligned; total ~44 MB)
    char* ws = (char*)d_ws;
    __bf16* Wbig  = (__bf16*)(ws);              // 4096*3072 bf16
    __bf16* Whh_b = (__bf16*)(ws + 25165824);   // 4096*1024 bf16
    __bf16* aWh   = (__bf16*)(ws + 33554432);   // 512*1024  bf16
    __bf16* xbf   = (__bf16*)(ws + 34603008);   // 256*3072  bf16
    __bf16* hbf   = (__bf16*)(ws + 36175872);   // 256*1024  bf16
    float*  gates = (float*)(ws + 36700160);    // 256*4096  f32
    float*  atth  = (float*)(ws + 40894464);    // 256*512   f32
    float*  scores= (float*)(ws + 41418752);    // 256*196   f32
    float*  wts   = (float*)(ws + 41619456);    // 256*196   f32
    float*  attA  = (float*)(ws + 41820160);    // 256*1024  f32
    float*  attB  = (float*)(ws + 42868736);    // 256*1024  f32

    auto cvt = [&](__bf16* dst, int dld, const float* src, int sld, int rows, int cols) {
        int total = rows * cols;
        cvt_slice_k<<<(total + 255) / 256, 256, 0, stream>>>(dst, dld, src, sld, rows, cols);
    };
    auto gemm = [&](const __bf16* A1, const __bf16* W1, int K1,
                    const __bf16* A2, const __bf16* W2, int K2,
                    const float* b1, const float* b2, float* C, int N) {
        dim3 g(N / 256, Bn / 64);
        gemm_wmma_k<<<g, 256, 0, stream>>>(A1, W1, K1, A2, W2, K2, b1, b2, C, N);
    };
    auto lstm = [&](const float* cprev, float* hout, float* cout, float* hout2) {
        lstm_pointwise_k<<<(Bn * Rn) / 256, 256, 0, stream>>>(gates, cprev, hout, cout, hout2);
    };
    auto attention = [&](const float* Wh, const float* bh, const float* Wa, const float* ba,
                         const float* p, const float* m, const float* feats,
                         const float* h_f32, float* out_att) {
        cvt(aWh, Rn, Wh, Rn, Hn, Rn);
        cvt(hbf, Rn, h_f32, Rn, Bn, Rn);
        gemm(hbf, aWh, Rn, nullptr, nullptr, 0, bh, nullptr, atth, Hn);
        att_scores_k<<<(Bn * Ln) / 8, 256, 0, stream>>>(p, atth, Wa, ba, scores);
        softmax_mask_k<<<Bn, 256, 0, stream>>>(scores, m, wts);
        wsum_k<<<dim3(Rn / 256, Bn), 256, 0, stream>>>(wts, feats, out_att);
    };

    // ---- x0 = [prev_h, fc_feats[:, :R], xt] ----
    cvt(xbf,        3072, state_h + 2 * BR, Rn,     Bn, Rn);
    cvt(xbf + 1024, 3072, fc,               3 * Rn, Bn, Rn);
    cvt(xbf + 2048, 3072, xt,               En,     Bn, En);

    // ---- LSTM cell 0 ----
    cvt(Wbig,  3072, Wih0, 3072, 4096, 3072);
    cvt(Whh_b, 1024, Whh0, 1024, 4096, 1024);
    cvt(hbf, Rn, state_h, Rn, Bn, Rn);
    gemm(xbf, Wbig, 3072, hbf, Whh_b, 1024, bih0, bhh0, gates, 4096);
    lstm(state_c, out_h, out_c, nullptr);                 // h_att0, c_att0

    // ---- att0 over (att_feats1, p_att_feats1, masks1) ----
    attention(a0_Wh, a0_bh, a0_Wa, a0_ba, p_att1, mask1, att_feats1, out_h, attA);

    // ---- x1 = [prev_h, att0, xt]: only middle slice changes ----
    cvt(xbf + 1024, 3072, attA, Rn, Bn, Rn);

    // ---- LSTM cell 1 ----
    cvt(Wbig,  3072, Wih1, 3072, 4096, 3072);
    cvt(Whh_b, 1024, Whh1, 1024, 4096, 1024);
    cvt(hbf, Rn, state_h + BR, Rn, Bn, Rn);
    gemm(xbf, Wbig, 3072, hbf, Whh_b, 1024, bih1, bhh1, gates, 4096);
    lstm(state_c + BR, out_h + BR, out_c + BR, nullptr);  // h_att, c_att

    // ---- att / att1 from h_att ----
    attention(a_Wh,  a_bh,  a_Wa,  a_ba,  p_att,  mask,  att_feats,  out_h + BR, attA);
    attention(a1_Wh, a1_bh, a1_Wa, a1_ba, p_att1, mask1, att_feats1, out_h + BR, attB);

    // ---- xL = [att, att1, h_att] ----
    cvt(xbf,        3072, attA,        Rn, Bn, Rn);
    cvt(xbf + 1024, 3072, attB,        Rn, Bn, Rn);
    cvt(xbf + 2048, 3072, out_h + BR,  Rn, Bn, Rn);

    // ---- LSTM lang cell ----
    cvt(Wbig,  3072, WihL, 3072, 4096, 3072);
    cvt(Whh_b, 1024, WhhL, 1024, 4096, 1024);
    cvt(hbf, Rn, state_h + 2 * BR, Rn, Bn, Rn);
    gemm(xbf, Wbig, 3072, hbf, Whh_b, 1024, bihL, bhhL, gates, 4096);
    lstm(state_c + 2 * BR, out_h + 2 * BR, out_c + 2 * BR, out); // h_lang + copy to output
}
